// CrossSparseWindowAttention_77103252897930
// MI455X (gfx1250) — matrix operand order repaired and checked by reading, not compiled, not run on hardware
//
#include <hip/hip_runtime.h>
#include <hip/hip_bf16.h>

typedef __attribute__((ext_vector_type(16))) _Float16 v16h;
typedef __attribute__((ext_vector_type(8)))  _Float16 v8h;
typedef __attribute__((ext_vector_type(8)))  float    v8f;

// ---------------- LDS layout (bytes, all 16B aligned) ----------------
#define STR96   104   // halves (K=96 matrices)
#define STR_VT  72    // halves (K=64 tokens)
#define STR_SIM 68    // floats
#define STR_P   72    // halves

#define OFF_WQKVT 0                         // [288][104] f16  = 59904
#define OFF_WOUTT (OFF_WQKVT + 59904)       // [96][104]  f16  = 19968
#define OFF_X     (OFF_WOUTT + 19968)       // [64][104]  f16  = 13312
#define OFF_Q     (OFF_X     + 13312)       // [64][104]  f16  = 13312
#define OFF_K     (OFF_Q     + 13312)       // [64][104]  f16  = 13312
#define OFF_VT    (OFF_K     + 13312)       // [96][72]   f16  = 13824
#define OFF_SIM   (OFF_VT    + 13824)       // [64][68]   f32  = 17408
#define OFF_P     (OFF_SIM   + 17408)       // [64][72]   f16  =  9216
#define OFF_OATT  (OFF_P     + 9216)        // [64][104]  f16  = 13312
#define OFF_BT    (OFF_OATT  + 13312)       // [225*3]    f32  ->  2720
#define OFF_BQKV  (OFF_BT    + 2720)        // [288]      f32  =  1152
#define OFF_BOUT  (OFF_BQKV  + 1152)        // [96]       f32  ->   384
#define OFF_GT    (OFF_BOUT  + 384)         // [2][96]    f32  =   768
#define OFF_KG    (OFF_GT    + 768)         // [2][96]    f32  =   768
#define OFF_VG    (OFF_KG    + 768)         // [2][96]    f32  =   768
#define OFF_PG    (OFF_VG    + 768)         // [64][2]    f32  =   512
#define SMEM_TOTAL (OFF_PG + 512)           // = 180640 bytes (< 320KB/WGP)

__device__ __forceinline__ v8f wmma16(v16h a, v16h b, v8f c) {
  return __builtin_amdgcn_wmma_f32_16x16x32_f16(false, a, false, b,
                                                (short)0, c, false, false);
}

// A fragment: row-major [M][K] f16 in LDS. 16-bit A 16x32 layout:
// lane m = l&15, kh = l>>4 ; elems 0..7 -> K=k0+kh*8+0..7 ; elems 8..15 -> K=k0+16+kh*8+0..7
__device__ __forceinline__ v16h load_frag_a(const _Float16* base, int stride,
                                            int m0, int k0, int lane) {
  int m = lane & 15, kh = lane >> 4;
  const _Float16* row = base + (m0 + m) * stride + k0 + kh * 8;
  union { v16h v; v8h h[2]; } u;
  u.h[0] = *(const v8h*)(row);
  u.h[1] = *(const v8h*)(row + 16);
  return u.v;
}

// B fragment from transposed storage Bt = [N][K] f16 in LDS.
// lane n = l&15, kh = l>>4 ; elems 0..7 -> K=k0+kh*8+0..7 ; elems 8..15 -> K=k0+16+kh*8+0..7
__device__ __forceinline__ v16h load_frag_b(const _Float16* base, int stride,
                                            int n0, int k0, int lane) {
  int n = lane & 15, kh = lane >> 4;
  const _Float16* row = base + (n0 + n) * stride + k0 + kh * 8;
  union { v16h v; v8h h[2]; } u;
  u.h[0] = *(const v8h*)(row);
  u.h[1] = *(const v8h*)(row + 16);
  return u.v;
}

__global__ __launch_bounds__(256, 1)
void swin_fused_wmma_kernel(const float* __restrict__ x,
                            const float* __restrict__ w_qkv,
                            const float* __restrict__ b_qkv,
                            const float* __restrict__ gtok,
                            const float* __restrict__ w_gkv,
                            const float* __restrict__ bias_tab,
                            const float* __restrict__ w_out,
                            const float* __restrict__ b_out,
                            float* __restrict__ out)
{
  constexpr int C = 96, MM = 64;
  const int tid  = threadIdx.x;
  const int lane = tid & 31;
  // wave index is uniform across the wave: pin it to an SGPR so all tile
  // indexing / Q-K-V selection compiles to SALU + s_cbranch (no EXEC masking).
  const int wave = __builtin_amdgcn_readfirstlane(tid >> 5);
  const int nw = blockIdx.x, nh = blockIdx.y, bidx = blockIdx.z;

  extern __shared__ char smem[];
  _Float16* sWqkvT = (_Float16*)(smem + OFF_WQKVT);
  _Float16* sWoutT = (_Float16*)(smem + OFF_WOUTT);
  _Float16* sX     = (_Float16*)(smem + OFF_X);
  _Float16* sQ     = (_Float16*)(smem + OFF_Q);
  _Float16* sK     = (_Float16*)(smem + OFF_K);
  _Float16* sVT    = (_Float16*)(smem + OFF_VT);
  float*    sSim   = (float*)   (smem + OFF_SIM);
  _Float16* sP     = (_Float16*)(smem + OFF_P);
  _Float16* sOatt  = (_Float16*)(smem + OFF_OATT);
  float*    sBT    = (float*)   (smem + OFF_BT);
  float*    sBqkv  = (float*)   (smem + OFF_BQKV);
  float*    sBout  = (float*)   (smem + OFF_BOUT);
  float*    sGT    = (float*)   (smem + OFF_GT);
  float*    sKG    = (float*)   (smem + OFF_KG);
  float*    sVG    = (float*)   (smem + OFF_VG);
  float*    sPG    = (float*)   (smem + OFF_PG);

  // ---- Phase A: stage weights (f32->f16, transposed) + x window ----
  for (int i = tid; i < 96 * 288; i += 256) {
    int c = i / 288, n = i % 288;
    sWqkvT[n * STR96 + c] = (_Float16)w_qkv[i];
  }
  for (int i = tid; i < 96 * 96; i += 256) {
    int c = i / 96, n = i % 96;
    sWoutT[n * STR96 + c] = (_Float16)w_out[i];
  }
  for (int i = tid; i < 288; i += 256) sBqkv[i] = b_qkv[i];
  if (tid < 96) sBout[tid] = b_out[tid];
  for (int i = tid; i < 225 * 3; i += 256) sBT[i] = bias_tab[i];
  for (int i = tid; i < MM * C; i += 256) {
    int p = i / C, c = i % C;
    int mi = p >> 3, mj = p & 7;
    size_t g = ((((size_t)bidx * 256) + nh * 8 + mi) * 256 + (nw * 8 + mj)) * 96 + c;
    sX[p * STR96 + c] = (_Float16)x[g];
  }
  __syncthreads();

  // ---- Phase B: x_mean + global tokens, QKV GEMM (64x96 @ 96x288) ----
  if (tid < C) {
    float s = 0.f;
    for (int p = 0; p < MM; ++p) s += (float)sX[p * STR96 + tid];
    float mean = s * (1.0f / 64.0f);
    sGT[tid]      = gtok[tid]      + mean;
    sGT[96 + tid] = gtok[96 + tid] + mean;
  }
  // 72 tiles / 8 waves = 9 per wave (constant trip count -> unrolled)
#pragma unroll
  for (int it = 0; it < 9; ++it) {
    int t = wave + it * 8;            // scalar
    int mt = t / 18, nt = t % 18;     // scalar
    v8f acc = {};
#pragma unroll
    for (int kt = 0; kt < 3; ++kt) {
      v16h a = load_frag_a(sX, STR96, mt * 16, kt * 32, lane);
      v16h b = load_frag_b(sWqkvT, STR96, nt * 16, kt * 32, lane);
      acc = wmma16(a, b, acc);
    }
    int n = lane & 15, hi = lane >> 4;
    float bj = sBqkv[nt * 16 + n];
    // nt is scalar: clean s_cbranch, no EXEC divergence.
    if (nt < 6) {                     // -> Q [token][96]
      int j = nt * 16 + n;
#pragma unroll
      for (int v = 0; v < 8; ++v) {
        int m = mt * 16 + v + 8 * hi;
        sQ[m * STR96 + j] = (_Float16)(acc[v] + bj);
      }
    } else if (nt < 12) {             // -> K [token][96]
      int j = (nt - 6) * 16 + n;
#pragma unroll
      for (int v = 0; v < 8; ++v) {
        int m = mt * 16 + v + 8 * hi;
        sK[m * STR96 + j] = (_Float16)(acc[v] + bj);
      }
    } else {                          // -> V transposed [96][64] (contiguous b128 store)
      int r = (nt - 12) * 16 + n;
#pragma unroll
      for (int v = 0; v < 8; ++v) {
        int m = mt * 16 + v + 8 * hi;
        sVT[r * STR_VT + m] = (_Float16)(acc[v] + bj);
      }
    }
  }
  __syncthreads();

  // ---- Phase C: global k/v projection: gkv = gt @ w_gkv (small, VALU) ----
  if (tid < 192) {
    float a0 = 0.f, a1 = 0.f;
    for (int c = 0; c < 96; ++c) {
      float w = w_gkv[c * 192 + tid];
      a0 += sGT[c] * w;
      a1 += sGT[96 + c] * w;
    }
    if (tid < 96) { sKG[tid] = a0;       sKG[96 + tid] = a1; }
    else          { sVG[tid - 96] = a0;  sVG[96 + tid - 96] = a1; }
  }
  __syncthreads();

  // ---- Phase D: per-head attention ----
  const float scale = 0.17677669529663687f;  // 1/sqrt(32)
  for (int h = 0; h < 3; ++h) {
    // sim = scale * Q Kt + rel_bias ; hd=32 -> one WMMA per 16x16 tile; 2 tiles/wave
#pragma unroll
    for (int it = 0; it < 2; ++it) {
      int t = wave + it * 8;          // scalar
      int mt = t >> 2, nt = t & 3;    // scalar
      v16h a = load_frag_a(sQ, STR96, mt * 16, h * 32, lane);
      v16h b = load_frag_b(sK, STR96, nt * 16, h * 32, lane);
      v8f acc = {};
      acc = wmma16(a, b, acc);
      int n = lane & 15, hi = lane >> 4;
      int q_abs = nt * 16 + n;
      int qi = q_abs >> 3, qj = q_abs & 7;
#pragma unroll
      for (int v = 0; v < 8; ++v) {
        int p_abs = mt * 16 + v + 8 * hi;
        int pi = p_abs >> 3, pj = p_abs & 7;
        int ridx = (pi - qi + 7) * 15 + (pj - qj + 7);
        sSim[p_abs * STR_SIM + q_abs] = scale * acc[v] + sBT[ridx * 3 + h];
      }
    }
    // global sim (UNSCALED q per reference) + 2-way softmax
    if (tid < 64) {
      float s0 = 0.f, s1 = 0.f;
      for (int k = 0; k < 32; ++k) {
        float qv = (float)sQ[tid * STR96 + h * 32 + k];
        s0 += qv * sKG[h * 32 + k];
        s1 += qv * sKG[96 + h * 32 + k];
      }
      float mx = fmaxf(s0, s1);
      float e0 = __expf(s0 - mx), e1 = __expf(s1 - mx);
      float inv = 1.0f / (e0 + e1);
      sPG[tid * 2]     = e0 * inv;
      sPG[tid * 2 + 1] = e1 * inv;
    }
    __syncthreads();
    // local softmax rows; fold (1-alpha)=0.75 into P
    if (tid < 64) {
      const float* row = sSim + tid * STR_SIM;
      float mx = row[0];
      for (int q = 1; q < 64; ++q) mx = fmaxf(mx, row[q]);
      float ssum = 0.f;
      for (int q = 0; q < 64; ++q) ssum += __expf(row[q] - mx);
      float inv = 0.75f / ssum;
      for (int q = 0; q < 64; ++q)
        sP[tid * STR_P + q] = (_Float16)(__expf(row[q] - mx) * inv);
    }
    __syncthreads();
    // out = P @ V (+ 0.25 * global path); 8 tiles -> 1 per wave
    {
      int mt = wave >> 1, nt = wave & 1;   // scalar
      v8f acc = {};
#pragma unroll
      for (int kt = 0; kt < 2; ++kt) {
        v16h a = load_frag_a(sP, STR_P, mt * 16, kt * 32, lane);
        v16h b = load_frag_b(sVT, STR_VT, h * 32 + nt * 16, kt * 32, lane);
        acc = wmma16(a, b, acc);
      }
      int n = lane & 15, hi = lane >> 4;
      int ch = h * 32 + nt * 16 + n;
#pragma unroll
      for (int v = 0; v < 8; ++v) {
        int p_abs = mt * 16 + v + 8 * hi;
        float g = 0.25f * (sPG[p_abs * 2] * sVG[ch] +
                           sPG[p_abs * 2 + 1] * sVG[96 + ch]);
        sOatt[p_abs * STR96 + ch] = (_Float16)(acc[v] + g);
      }
    }
    __syncthreads();
  }

  // ---- Phase E: output projection [64,96] @ [96,96] + b_out -> global ----
#pragma unroll
  for (int it = 0; it < 3; ++it) {
    int t = wave + it * 8;            // scalar
    int mt = t / 6, nt = t % 6;       // scalar
    v8f acc = {};
#pragma unroll
    for (int kt = 0; kt < 3; ++kt) {
      v16h a = load_frag_a(sOatt, STR96, mt * 16, kt * 32, lane);
      v16h b = load_frag_b(sWoutT, STR96, nt * 16, kt * 32, lane);
      acc = wmma16(a, b, acc);
    }
    int n = lane & 15, hi = lane >> 4;
    int c = nt * 16 + n;
    float bo = sBout[c];
#pragma unroll
    for (int v = 0; v < 8; ++v) {
      int p = mt * 16 + v + 8 * hi;
      int mi = p >> 3, mj = p & 7;
      size_t g = ((((size_t)bidx * 256) + nh * 8 + mi) * 256 + (nw * 8 + mj)) * 96 + c;
      out[g] = acc[v] + bo;
    }
  }
}

extern "C" void kernel_launch(void* const* d_in, const int* in_sizes, int n_in,
                              void* d_out, int out_size, void* d_ws, size_t ws_size,
                              hipStream_t stream) {
  (void)in_sizes; (void)n_in; (void)d_ws; (void)ws_size; (void)out_size;
  const float* x        = (const float*)d_in[0];
  const float* w_qkv    = (const float*)d_in[1];
  const float* b_qkv    = (const float*)d_in[2];
  const float* gtok     = (const float*)d_in[3];
  const float* w_gkv    = (const float*)d_in[4];
  const float* bias_tab = (const float*)d_in[5];
  const float* w_out    = (const float*)d_in[6];
  const float* b_out    = (const float*)d_in[7];
  float* out = (float*)d_out;

  dim3 grid(32, 32, 8);   // (nW, nH, B) -> 8192 windows
  dim3 block(256);        // 8 waves (wave32)
  swin_fused_wmma_kernel<<<grid, block, SMEM_TOTAL, stream>>>(
      x, w_qkv, b_qkv, gtok, w_gkv, bias_tab, w_out, b_out, out);
}
